// GatedDeltaNetLayer_31920196944236
// MI455X (gfx1250) — matrix operand order repaired and checked by reading, not compiled, not run on hardware
//
#include <hip/hip_runtime.h>

#define S_LEN 4096
#define HID   2048
#define NH    8
#define NVH   16
#define DKK   128
#define DVV   128
#define KEY_DIM 1024
#define VAL_DIM 2048

typedef __attribute__((ext_vector_type(16))) __bf16 v16bf;
typedef __attribute__((ext_vector_type(8)))  float  v8f;

static __device__ __forceinline__ v8f wmma_bf16(v16bf a, v16bf b, v8f c) {
  return __builtin_amdgcn_wmma_f32_16x16x32_bf16(false, a, false, b, (short)0, c,
                                                 false, false);
}

// ---------------------------------------------------------------------------
// gfx1250 async global->LDS DMA (ASYNCcnt path), with safe fallback
// Builtin signature (from hipcc diagnostic): arg0 = int4 AS(1)*, 4 args total.
// ---------------------------------------------------------------------------
typedef int v4i_gcc __attribute__((vector_size(16)));

#if defined(__has_builtin)
#if __has_builtin(__builtin_amdgcn_global_load_async_to_lds_b128)
#define HAVE_ASYNC_LDS 1
#endif
#endif
#ifndef HAVE_ASYNC_LDS
#define HAVE_ASYNC_LDS 0
#endif

static __device__ __forceinline__ void async_copy16(const void* g, void* l) {
#if HAVE_ASYNC_LDS
  __builtin_amdgcn_global_load_async_to_lds_b128(
      (__attribute__((address_space(1))) v4i_gcc*)(uintptr_t)(g),
      (__attribute__((address_space(3))) v4i_gcc*)(uint32_t)(uintptr_t)(l),
      0, 0);
#else
  *(uint4*)l = *(const uint4*)g;
#endif
}

static __device__ __forceinline__ void async_wait_all() {
#if HAVE_ASYNC_LDS
#if __has_builtin(__builtin_amdgcn_s_wait_asynccnt)
  __builtin_amdgcn_s_wait_asynccnt(0);
#else
  asm volatile("s_wait_asynccnt 0x0" ::: "memory");
#endif
#endif
}

// ---------------------------------------------------------------------------
// f32 -> bf16 conversion
// ---------------------------------------------------------------------------
__global__ void cvt_bf16(const float* __restrict__ s, __bf16* __restrict__ d, int n) {
  int i = blockIdx.x * blockDim.x + threadIdx.x;
  if (i < n) d[i] = (__bf16)s[i];
}

// ---------------------------------------------------------------------------
// Y[M=4096][N] = X[M][K] @ W[N][K]^T   (bf16 in, f32 out, WMMA bf16)
// 256 thr (8 waves as 4x2), tile 128x64, K-panel 32,
// double-buffered LDS fed by async global->LDS DMA.
// ---------------------------------------------------------------------------
__global__ __launch_bounds__(256)
void gemm_bf16_xt(const __bf16* __restrict__ X, const __bf16* __restrict__ W,
                  float* __restrict__ Y, int N, int K) {
  __shared__ alignas(16) __bf16 sA[2][128 * 40];
  __shared__ alignas(16) __bf16 sB[2][64 * 40];
  const int tid = threadIdx.x, lane = tid & 31, wv = tid >> 5;
  const int m0 = blockIdx.y * 128, n0 = blockIdx.x * 64;
  const int wm = (wv & 3) * 32, wn = (wv >> 2) * 32;
  v8f acc[2][2] = {};

  auto stage = [&](int buf, int kb) {
    for (int v = tid; v < 512; v += 256) {
      int r = v >> 2, kc = (v & 3) * 8;
      async_copy16(&X[(size_t)(m0 + r) * K + kb + kc], &sA[buf][r * 40 + kc]);
    }
    {
      int r = tid >> 2, kc = (tid & 3) * 8;
      int n = n0 + r;
      if (n < N) async_copy16(&W[(size_t)n * K + kb + kc], &sB[buf][r * 40 + kc]);
    }
  };

  stage(0, 0);
  for (int kb = 0; kb < K; kb += 32) {
    const int cur = (kb >> 5) & 1;
    async_wait_all();
    __syncthreads();
    if (kb + 32 < K) stage(cur ^ 1, kb + 32);

    v16bf af[2], bfr[2];
#pragma unroll
    for (int i = 0; i < 2; ++i) {
      int m = wm + i * 16 + (lane & 15);
      int ka = (lane & 16) ? 8 : 0;
      v16bf a;
#pragma unroll
      for (int e = 0; e < 8; ++e) {
        a[e]     = sA[cur][m * 40 + ka + e];
        a[e + 8] = sA[cur][m * 40 + ka + 16 + e];
      }
      af[i] = a;
      int n = wn + i * 16 + (lane & 15);
      int kB = (lane & 16) ? 8 : 0;
      v16bf b;
#pragma unroll
      for (int e = 0; e < 16; ++e) b[e] = sB[cur][n * 40 + kB + (e & 7) + ((e >> 3) << 4)];
      bfr[i] = b;
    }
#pragma unroll
    for (int i = 0; i < 2; ++i)
#pragma unroll
      for (int j = 0; j < 2; ++j) acc[i][j] = wmma_bf16(af[i], bfr[j], acc[i][j]);
  }
#pragma unroll
  for (int i = 0; i < 2; ++i)
#pragma unroll
    for (int j = 0; j < 2; ++j) {
      int mb = m0 + wm + i * 16 + ((lane & 16) ? 8 : 0);
      int n  = n0 + wn + j * 16 + (lane & 15);
      if (n < N) {
#pragma unroll
        for (int r = 0; r < 8; ++r) Y[(size_t)(mb + r) * N + n] = acc[i][j][r];
      }
    }
}

// ---------------------------------------------------------------------------
// causal depthwise conv (K=4) + SiLU
// ---------------------------------------------------------------------------
__global__ void conv_silu(const float* __restrict__ pre, const float* __restrict__ w,
                          float* __restrict__ out, int C) {
  int i = blockIdx.x * blockDim.x + threadIdx.x;
  if (i >= S_LEN * C) return;
  int t = i / C, c = i - t * C;
  float acc = 0.f;
#pragma unroll
  for (int j = 0; j < 4; ++j) {
    int ts = t - 3 + j;
    if (ts >= 0) acc += pre[(size_t)ts * C + c] * w[c * 4 + j];
  }
  out[i] = acc / (1.f + __expf(-acc));
}

// ---------------------------------------------------------------------------
// per-(t, kq-head) L2 normalize over 128 dims (+ optional SCALE)
// ---------------------------------------------------------------------------
__global__ void l2norm_scale(float* __restrict__ buf, float scale) {
  int b = blockIdx.x;
  int t = b >> 3, hh = b & 7;
  int i = threadIdx.x;
  float* p = buf + (size_t)t * KEY_DIM + hh * DKK;
  __shared__ float red[128];
  float v = p[i];
  red[i] = v * v;
  __syncthreads();
  for (int s = 64; s > 0; s >>= 1) {
    if (i < s) red[i] += red[i + s];
    __syncthreads();
  }
  p[i] = v * rsqrtf(red[0] + 1e-6f) * scale;
}

// ---------------------------------------------------------------------------
// g = -exp(A_log)*softplus(a+dt_bias), beta = sigmoid(b)
// ---------------------------------------------------------------------------
__global__ void prep_gb(const float* __restrict__ ab, const float* __restrict__ Alog,
                        const float* __restrict__ dtb, float* __restrict__ g,
                        float* __restrict__ beta) {
  int i = blockIdx.x * blockDim.x + threadIdx.x;
  if (i >= S_LEN * NVH) return;
  int t = i >> 4, hh = i & 15;
  float a = ab[t * 32 + hh] + dtb[hh];
  float sp = (a > 20.f) ? a : log1pf(__expf(a));
  g[i] = -__expf(Alog[hh]) * sp;
  float b = ab[t * 32 + 16 + hh];
  beta[i] = 1.f / (1.f + __expf(-b));
}

// ---------------------------------------------------------------------------
// Chunked delta-rule scan: one workgroup (8 waves) per value head.
// Chunk C=64. Per chunk:
//   B: M  = strict-lower( exp(Gt-Gj) * K K^T ),  Lq = causal( exp(Gt-Gj) * Q K^T )
//   C: P  = diag(exp(G)) K @ S0 ;  RHS = beta*(V - P)
//   D: forward substitution D[t] = RHS[t] - beta_t * sum_{j<t} M[t,j] D[j]
//      (each lane owns one DV column -> no sync needed)
//   E: O  = diag(exp(G)) Q @ S0 + Lq @ D
//   F: S  = exp(G_C) * S0 + K2^T @ D,  K2[j,:] = exp(G_C - G_j) k_j
// K chunk arrives via async DMA overlapped with the decay prefix scan.
// ---------------------------------------------------------------------------
__global__ __launch_bounds__(256)
void scan_kernel(const float* __restrict__ Q, const float* __restrict__ Kc,
                 const float* __restrict__ V, const float* __restrict__ Gv,
                 const float* __restrict__ Bv, float* __restrict__ O) {
  const int h = blockIdx.x;   // value head
  const int kh = h >> 1;      // shared q/k head (GVA = 2)
  const int tid = threadIdx.x, lane = tid & 31, wv = tid >> 5;

  __shared__ float  sS[128 * 132];   // state, f32
  __shared__ float  sK[64 * 132];    // raw K chunk, f32
  __shared__ __bf16 sD[64 * 136];    // P -> RHS -> D
  __shared__ float  sM[64 * 66];     // strict-lower scaled KK^T
  __shared__ __bf16 sLq[64 * 72];    // causal scaled QK^T
  __shared__ float  sG[64], sB[64], sR2[64];

  for (int i = tid; i < 128 * 132; i += 256) sS[i] = 0.f;
  __syncthreads();

  for (int c = 0; c < 64; ++c) {
    const int t0 = c * 64;
    // ---- Phase A: async-DMA K chunk; decay prefix; prefetch Q/V ----
    for (int i = tid; i < 2048; i += 256) {        // 64 rows x 32 float4
      int t = i >> 5, d4 = (i & 31) * 4;
      async_copy16(&Kc[(size_t)(t0 + t) * KEY_DIM + kh * DKK + d4],
                   &sK[t * 132 + d4]);
    }
    if (tid < 128) {                               // warm L2/L0 for Q rows
      int t = tid >> 1, half = (tid & 1) * 64;
      __builtin_prefetch(&Q[(size_t)(t0 + t) * KEY_DIM + kh * DKK + half], 0, 1);
    } else {                                       // and V rows
      int i = tid - 128;
      int t = i >> 1, half = (i & 1) * 64;
      __builtin_prefetch(&V[(size_t)(t0 + t) * VAL_DIM + h * DVV + half], 0, 1);
    }
    if (tid < 64) {
      sG[tid] = Gv[(t0 + tid) * NVH + h];
      sB[tid] = Bv[(t0 + tid) * NVH + h];
    }
    __syncthreads();
    if (tid == 0) {
      float run = 0.f;
      for (int t = 0; t < 64; ++t) { run += sG[t]; sG[t] = run; }
    }
    __syncthreads();
    if (tid < 64) sR2[tid] = __expf(sG[63] - sG[tid]);
    async_wait_all();                              // K chunk landed in LDS
    __syncthreads();

    // ---- Phase B: KK^T -> sM, QK^T -> sLq (32 16x16 tiles) ----
    for (int job = wv * 4; job < wv * 4 + 4; ++job) {
      const bool isQ = job >= 16;
      const int tt = job & 15;
      const int ti = (tt >> 2) * 16, tj = (tt & 3) * 16;
      v8f acc = {};
      for (int ks = 0; ks < 4; ++ks) {
        const int k0 = ks * 32;
        v16bf a, b;
        {
          int m = ti + (lane & 15);
          int kb = k0 + ((lane & 16) ? 8 : 0);
          if (isQ) {
            const float* row = Q + (size_t)(t0 + m) * KEY_DIM + kh * DKK + kb;
#pragma unroll
            for (int e = 0; e < 8; ++e) { a[e] = (__bf16)row[e]; a[e + 8] = (__bf16)row[16 + e]; }
          } else {
            const float* row = sK + m * 132 + kb;
#pragma unroll
            for (int e = 0; e < 8; ++e) { a[e] = (__bf16)row[e]; a[e + 8] = (__bf16)row[16 + e]; }
          }
        }
        {
          int n = tj + (lane & 15);
          int kb = k0 + ((lane & 16) ? 8 : 0);
          const float* row = sK + n * 132 + kb;
#pragma unroll
          for (int e = 0; e < 16; ++e) b[e] = (__bf16)row[(e & 7) + ((e >> 3) << 4)];
        }
        acc = wmma_bf16(a, b, acc);
      }
      const int n = tj + (lane & 15);
      const int mb = ti + ((lane & 16) ? 8 : 0);
#pragma unroll
      for (int r = 0; r < 8; ++r) {
        const int m = mb + r;
        const float sc = __expf(sG[m] - sG[n]);
        if (isQ)
          sLq[m * 72 + n] = (__bf16)((n <= m) ? acc[r] * sc : 0.f);
        else
          sM[m * 66 + n] = (n < m) ? acc[r] * sc : 0.f;
      }
    }
    __syncthreads();

    // ---- Phase C: P = Kd @ S0 ; RHS = beta*(V - P) -> sD ----
    for (int job = wv * 4; job < wv * 4 + 4; ++job) {
      const int ti = (job >> 3) * 16, dj = (job & 7) * 16;
      v8f acc = {};
      for (int ks = 0; ks < 4; ++ks) {
        const int k0 = ks * 32;
        v16bf a, b;
        {
          int m = ti + (lane & 15);
          int kb = k0 + ((lane & 16) ? 8 : 0);
          float rm = __expf(sG[m]);
          const float* row = sK + m * 132 + kb;
#pragma unroll
          for (int e = 0; e < 8; ++e) {
            a[e]     = (__bf16)(row[e] * rm);
            a[e + 8] = (__bf16)(row[16 + e] * rm);
          }
        }
        {
          int n = dj + (lane & 15);
          int kb = k0 + ((lane & 16) ? 8 : 0);
#pragma unroll
          for (int e = 0; e < 16; ++e) b[e] = (__bf16)sS[(kb + (e & 7) + ((e >> 3) << 4)) * 132 + n];
        }
        acc = wmma_bf16(a, b, acc);
      }
      const int n = dj + (lane & 15);
      const int mb = ti + ((lane & 16) ? 8 : 0);
#pragma unroll
      for (int r = 0; r < 8; ++r) {
        const int m = mb + r;
        const float vv = V[(size_t)(t0 + m) * VAL_DIM + h * DVV + n];
        sD[m * 136 + n] = (__bf16)(sB[m] * (vv - acc[r]));
      }
    }
    __syncthreads();

    // ---- Phase D: triangular solve, one lane per DV column ----
    if (tid < 128) {
      const int dv = tid;
      for (int t = 1; t < 64; ++t) {
        float sum = 0.f;
        for (int j = 0; j < t; ++j) sum += sM[t * 66 + j] * (float)sD[j * 136 + dv];
        sD[t * 136 + dv] = (__bf16)((float)sD[t * 136 + dv] - sB[t] * sum);
      }
    }
    __syncthreads();

    // ---- Phase E: O = Qd @ S0 + Lq @ D -> global ----
    for (int job = wv * 4; job < wv * 4 + 4; ++job) {
      const int ti = (job >> 3) * 16, dj = (job & 7) * 16;
      v8f acc = {};
      for (int ks = 0; ks < 4; ++ks) {
        const int k0 = ks * 32;
        v16bf a, b;
        {
          int m = ti + (lane & 15);
          int kb = k0 + ((lane & 16) ? 8 : 0);
          float rm = __expf(sG[m]);
          const float* row = Q + (size_t)(t0 + m) * KEY_DIM + kh * DKK + kb;
#pragma unroll
          for (int e = 0; e < 8; ++e) {
            a[e]     = (__bf16)(row[e] * rm);
            a[e + 8] = (__bf16)(row[16 + e] * rm);
          }
        }
        {
          int n = dj + (lane & 15);
          int kb = k0 + ((lane & 16) ? 8 : 0);
#pragma unroll
          for (int e = 0; e < 16; ++e) b[e] = (__bf16)sS[(kb + (e & 7) + ((e >> 3) << 4)) * 132 + n];
        }
        acc = wmma_bf16(a, b, acc);
      }
      for (int ks = 0; ks < 2; ++ks) {
        const int j0 = ks * 32;
        v16bf a, b;
        {
          int m = ti + (lane & 15);
          int jb = j0 + ((lane & 16) ? 8 : 0);
          const __bf16* row = sLq + m * 72 + jb;
#pragma unroll
          for (int e = 0; e < 8; ++e) { a[e] = row[e]; a[e + 8] = row[16 + e]; }
        }
        {
          int n = dj + (lane & 15);
          int kb = j0 + ((lane & 16) ? 8 : 0);
#pragma unroll
          for (int e = 0; e < 16; ++e) b[e] = sD[(kb + (e & 7) + ((e >> 3) << 4)) * 136 + n];
        }
        acc = wmma_bf16(a, b, acc);
      }
      const int n = dj + (lane & 15);
      const int mb = ti + ((lane & 16) ? 8 : 0);
#pragma unroll
      for (int r = 0; r < 8; ++r)
        O[(size_t)(t0 + mb + r) * VAL_DIM + h * DVV + n] = acc[r];
    }
    __syncthreads();

    // ---- Phase F: S = exp(G_C)*S0 + K2^T @ D ----
    const float egc = __expf(sG[63]);
    for (int job = wv * 8; job < wv * 8 + 8; ++job) {
      const int di = (job >> 3) * 16, dj = (job & 7) * 16;
      const int n = dj + (lane & 15);
      const int mb = di + ((lane & 16) ? 8 : 0);
      v8f acc;
#pragma unroll
      for (int r = 0; r < 8; ++r) acc[r] = egc * sS[(mb + r) * 132 + n];
      for (int ks = 0; ks < 2; ++ks) {
        const int j0 = ks * 32;
        v16bf a, b;
        {
          int m = di + (lane & 15);
          int jb = j0 + ((lane & 16) ? 8 : 0);
#pragma unroll
          for (int e = 0; e < 8; ++e) {
            int j1 = jb + e, j2 = jb + 16 + e;
            a[e]     = (__bf16)(sK[j1 * 132 + m] * sR2[j1]);
            a[e + 8] = (__bf16)(sK[j2 * 132 + m] * sR2[j2]);
          }
        }
        {
          int kb = j0 + ((lane & 16) ? 8 : 0);
#pragma unroll
          for (int e = 0; e < 16; ++e) b[e] = sD[(kb + (e & 7) + ((e >> 3) << 4)) * 136 + n];
        }
        acc = wmma_bf16(a, b, acc);
      }
#pragma unroll
      for (int r = 0; r < 8; ++r) sS[(mb + r) * 132 + n] = acc[r];
    }
    __syncthreads();
  }
}

// ---------------------------------------------------------------------------
// og = o * silu(gate); RMS-normalize over DV; * w_norm; -> bf16
// ---------------------------------------------------------------------------
__global__ void gate_norm(const float* __restrict__ Ob, const float* __restrict__ Gt,
                          const float* __restrict__ wn, __bf16* __restrict__ out) {
  int b = blockIdx.x;
  int t = b >> 4, hh = b & 15;
  int i = threadIdx.x;
  size_t off = (size_t)t * VAL_DIM + hh * DVV + i;
  float o = Ob[off], g = Gt[off];
  float og = o * (g / (1.f + __expf(-g)));
  __shared__ float red[128];
  red[i] = og * og;
  __syncthreads();
  for (int s = 64; s > 0; s >>= 1) {
    if (i < s) red[i] += red[i + s];
    __syncthreads();
  }
  out[off] = (__bf16)(og * rsqrtf(red[0] * (1.f / 128.f) + 1e-6f) * wn[i]);
}

// ---------------------------------------------------------------------------
extern "C" void kernel_launch(void* const* d_in, const int* /*in_sizes*/, int /*n_in*/,
                              void* d_out, int /*out_size*/, void* d_ws,
                              size_t /*ws_size*/, hipStream_t stream) {
  const float* x    = (const float*)d_in[0];
  const float* Wq   = (const float*)d_in[1];
  const float* Wk   = (const float*)d_in[2];
  const float* Wv   = (const float*)d_in[3];
  const float* Wa   = (const float*)d_in[4];
  const float* Wb   = (const float*)d_in[5];
  const float* Wg   = (const float*)d_in[6];
  const float* Wo   = (const float*)d_in[7];
  const float* cq   = (const float*)d_in[8];
  const float* ck   = (const float*)d_in[9];
  const float* cv   = (const float*)d_in[10];
  const float* Alog = (const float*)d_in[11];
  const float* dtb  = (const float*)d_in[12];
  const float* wn   = (const float*)d_in[13];
  float* out = (float*)d_out;

  char* p = (char*)d_ws;
  auto take = [&](size_t bytes) {
    char* r = p;
    p += (bytes + 255) & ~(size_t)255;
    return r;
  };

  __bf16* XB  = (__bf16*)take((size_t)S_LEN * HID * 2);
  __bf16* WqB = (__bf16*)take((size_t)KEY_DIM * HID * 2);
  __bf16* WkB = (__bf16*)take((size_t)KEY_DIM * HID * 2);
  __bf16* WvB = (__bf16*)take((size_t)VAL_DIM * HID * 2);
  __bf16* ABB = (__bf16*)take((size_t)32 * HID * 2);
  __bf16* WgB = (__bf16*)take((size_t)VAL_DIM * HID * 2);
  __bf16* WoB = (__bf16*)take((size_t)HID * VAL_DIM * 2);
  float* QP  = (float*)take((size_t)S_LEN * KEY_DIM * 4);
  float* KP  = (float*)take((size_t)S_LEN * KEY_DIM * 4);
  float* VP  = (float*)take((size_t)S_LEN * VAL_DIM * 4);
  float* ABP = (float*)take((size_t)S_LEN * 32 * 4);
  float* GT  = (float*)take((size_t)S_LEN * VAL_DIM * 4);
  float* Qb  = (float*)take((size_t)S_LEN * KEY_DIM * 4);
  float* Kb  = (float*)take((size_t)S_LEN * KEY_DIM * 4);
  float* Vb  = (float*)take((size_t)S_LEN * VAL_DIM * 4);
  float* Gd  = (float*)take((size_t)S_LEN * NVH * 4);
  float* Bt  = (float*)take((size_t)S_LEN * NVH * 4);
  float* Ob  = (float*)take((size_t)S_LEN * VAL_DIM * 4);
  __bf16* OGB = (__bf16*)take((size_t)S_LEN * VAL_DIM * 2);

  auto cvt = [&](const float* s, __bf16* d, size_t n) {
    cvt_bf16<<<(unsigned)((n + 255) / 256), 256, 0, stream>>>(s, d, (int)n);
  };
  cvt(x, XB, (size_t)S_LEN * HID);
  cvt(Wq, WqB, (size_t)KEY_DIM * HID);
  cvt(Wk, WkB, (size_t)KEY_DIM * HID);
  cvt(Wv, WvB, (size_t)VAL_DIM * HID);
  cvt(Wa, ABB, (size_t)NVH * HID);
  cvt(Wb, ABB + (size_t)NVH * HID, (size_t)NVH * HID);
  cvt(Wg, WgB, (size_t)VAL_DIM * HID);
  cvt(Wo, WoB, (size_t)HID * VAL_DIM);

  auto gemm = [&](const __bf16* A, const __bf16* W, float* Y, int N) {
    dim3 grid((N + 63) / 64, S_LEN / 128);
    gemm_bf16_xt<<<grid, 256, 0, stream>>>(A, W, Y, N, HID);
  };
  gemm(XB, WqB, QP, KEY_DIM);
  gemm(XB, WkB, KP, KEY_DIM);
  gemm(XB, WvB, VP, VAL_DIM);
  gemm(XB, ABB, ABP, 32);
  gemm(XB, WgB, GT, VAL_DIM);

  conv_silu<<<(S_LEN * KEY_DIM) / 256, 256, 0, stream>>>(QP, cq, Qb, KEY_DIM);
  conv_silu<<<(S_LEN * KEY_DIM) / 256, 256, 0, stream>>>(KP, ck, Kb, KEY_DIM);
  conv_silu<<<(S_LEN * VAL_DIM) / 256, 256, 0, stream>>>(VP, cv, Vb, VAL_DIM);

  l2norm_scale<<<S_LEN * NH, 128, 0, stream>>>(Qb, 0.08838834764831845f);  // DK^-0.5
  l2norm_scale<<<S_LEN * NH, 128, 0, stream>>>(Kb, 1.0f);

  prep_gb<<<(S_LEN * NVH + 255) / 256, 256, 0, stream>>>(ABP, Alog, dtb, Gd, Bt);

  scan_kernel<<<NVH, 256, 0, stream>>>(Qb, Kb, Vb, Gd, Bt, Ob);

  gate_norm<<<S_LEN * NVH, 128, 0, stream>>>(Ob, GT, wn, OGB);

  gemm(OGB, WoB, out, HID);
}
